// Transformer_54271206752578
// MI455X (gfx1250) — matrix operand order repaired and checked by reading, not compiled, not run on hardware
//
#include <hip/hip_runtime.h>
#include <hip/hip_bf16.h>
#include <cstddef>
#include <cstdint>

// ---------------- vector types ----------------
typedef _Float16 half16 __attribute__((ext_vector_type(16)));
typedef _Float16 half8  __attribute__((ext_vector_type(8)));
typedef float    float8 __attribute__((ext_vector_type(8)));
typedef unsigned int uint4v __attribute__((ext_vector_type(4)));
typedef unsigned int uint8v __attribute__((ext_vector_type(8)));

__device__ __forceinline__ half16 cat8(half8 a, half8 b) {
    return __builtin_shufflevector(a, b, 0,1,2,3,4,5,6,7,8,9,10,11,12,13,14,15);
}

__device__ __forceinline__ float8 wmma_f32(half16 a, half16 b, float8 c) {
    // v_wmma_f32_16x16x32_f16  (neg_a, A, neg_b, B, c_mod, C, reuse_a, reuse_b)
    return __builtin_amdgcn_wmma_f32_16x16x32_f16(false, a, false, b, (short)0, c, false, false);
}

// ---------------- CDNA5 async copy: global -> LDS (ASYNCcnt path) ----------------
__device__ __forceinline__ void async_b128(const void* g, void* lds) {
    unsigned           loff = (unsigned)(uintptr_t)lds;   // LDS byte offset (low 32 bits of generic ptr)
    unsigned long long ga   = (unsigned long long)(uintptr_t)g;
    asm volatile("global_load_async_to_lds_b128 %0, %1, off"
                 :: "v"(loff), "v"(ga) : "memory");
}

__device__ __forceinline__ void wait_async() {
#if __has_builtin(__builtin_amdgcn_s_wait_asynccnt)
    __builtin_amdgcn_s_wait_asynccnt(0);
#else
    asm volatile("s_wait_asynccnt 0x0" ::: "memory");
#endif
}

// ---------------- CDNA5 Tensor Data Mover: 2D tile global -> LDS (TENSORcnt path) ----------------
// D# per CDNA5 ISA ch.8 built in SGPR groups, issued via inline asm (portable across toolchains).
// Loads a 16-row x 32-half tile (row stride = rowStrideElems halfs) into LDS, packed row-major.
__device__ __forceinline__ void tdm_load_tile16x32(const void* gaddr, void* ldsaddr,
                                                   unsigned rowStrideElems) {
    unsigned long long ga = (unsigned long long)(uintptr_t)gaddr;
    unsigned           la = (unsigned)(uintptr_t)ldsaddr;
    uint4v g0;
    g0[0] = 1u;                                                  // count=1 (one valid descriptor)
    g0[1] = la;                                                  // lds_addr (bytes)
    g0[2] = (unsigned)ga;                                        // global_addr[31:0]
    g0[3] = (unsigned)((ga >> 32) & 0x01FFFFFFu) | (2u << 30);   // global_addr[56:32], type=2 ("image")
    uint8v g1;
    g1[0] = 1u << 16;             // workgroup_mask=0 (not in cluster), data_size=1 -> 2 bytes
    g1[1] = 32u << 16;            // tensor_dim0 = 32 (bits 79:48, low half)
    g1[2] = 16u << 16;            // tensor_dim0 hi = 0, tensor_dim1 = 16 (bits 111:80, low half)
    g1[3] = 32u << 16;            // tensor_dim1 hi = 0, tile_dim0 = 32 (bits 127:112)
    g1[4] = 16u;                  // tile_dim1 = 16, tile_dim2 = 0
    g1[5] = rowStrideElems;       // tensor_dim0_stride[31:0]
    g1[6] = 0u;                   // stride hi / dim1_stride lo
    g1[7] = 0u;
    uint4v gz = {0u, 0u, 0u, 0u}; // groups 2/3 unused (2D tensor)
    asm volatile("tensor_load_to_lds %0, %1, %2, %3"
                 :: "s"(g0), "s"(g1), "s"(gz), "s"(gz) : "memory");
}

__device__ __forceinline__ void wait_tensor() {
    asm volatile("s_wait_tensorcnt 0x0" ::: "memory");
}

// ---------------- fp32 -> fp16 weight conversion ----------------
__global__ void f32_to_f16_kernel(const float* __restrict__ src, _Float16* __restrict__ dst, int n) {
    int i = blockIdx.x * blockDim.x + threadIdx.x;
    if (i < n) dst[i] = (_Float16)src[i];
}

// ---------------- LayerNorm(x) -> f16 (row length 512, one wave per row) ----------------
__global__ __launch_bounds__(256) void ln_x_kernel(const float* __restrict__ x,
                                                   const float* __restrict__ g,
                                                   const float* __restrict__ b,
                                                   _Float16* __restrict__ h16, int M) {
    int row  = blockIdx.x * 8 + (threadIdx.x >> 5);
    int lane = threadIdx.x & 31;
    if (row >= M) return;
    const float* xr = x + (size_t)row * 512;
    float v[16];
    float s = 0.f;
#pragma unroll
    for (int i = 0; i < 16; ++i) { v[i] = xr[lane + 32 * i]; s += v[i]; }
#pragma unroll
    for (int m = 16; m >= 1; m >>= 1) s += __shfl_xor(s, m, 32);
    float mu = s * (1.f / 512.f);
    float var = 0.f;
#pragma unroll
    for (int i = 0; i < 16; ++i) { float d = v[i] - mu; var += d * d; }
#pragma unroll
    for (int m = 16; m >= 1; m >>= 1) var += __shfl_xor(var, m, 32);
    float rstd = rsqrtf(var * (1.f / 512.f) + 1e-5f);
#pragma unroll
    for (int i = 0; i < 16; ++i) {
        int c = lane + 32 * i;
        float y = (v[i] - mu) * rstd * g[c] + b[c];
        h16[(size_t)row * 512 + c] = (_Float16)y;
    }
}

// ---------------- WMMA GEMM, TDM-staged W tiles + async-staged A tiles ----------------
// C[M,N] = A[M,K] @ W[N,K]^T.  block: 256 threads = 8 waves (4M x 2N),
// block tile 128x128, wave tile 32x64, k-step 32, double-buffered LDS.
template <int K>
__global__ __launch_bounds__(256) void wmma_gemm_kernel(const _Float16* __restrict__ A,
                                                        const _Float16* __restrict__ W,
                                                        float* __restrict__ C, int N) {
    __shared__ __align__(64) _Float16 Abuf[2][128 * 32];
    __shared__ __align__(64) _Float16 Bbuf[2][128 * 32];

    int tid  = threadIdx.x;
    int wave = tid >> 5;
    int lane = tid & 31;
    int wm = wave & 3, wn = wave >> 2;
    int rowBase = blockIdx.x * 128;
    int colBase = blockIdx.y * 128;
    int l16 = lane & 15, hi = lane >> 4;

    // cooperative async copy mapping: thread t -> tile row t>>1, 16 contiguous halfs at (t&1)*16
    int ldRow = tid >> 1;
    int ldOff = (tid & 1) * 16;

    const _Float16* gA = A + (size_t)(rowBase + ldRow) * K + ldOff;
    _Float16* lA = &Abuf[0][ldRow * 32 + ldOff];   // buffer stride = 128*32 halfs
    const _Float16* gWw = W + (size_t)(colBase + wave * 16) * K;  // this wave's 16 weight rows

    auto issue = [&](int buf, int kk) {
        size_t lofs = (size_t)buf * 128 * 32;
        async_b128(gA + kk,     lA + lofs);
        async_b128(gA + kk + 8, lA + lofs + 8);
        // one TDM descriptor per wave: 16 rows x 32 halfs, row stride K
        tdm_load_tile16x32(gWw + kk, &Bbuf[buf][wave * 16 * 32], (unsigned)K);
    };

    issue(0, 0);
    wait_async();
    wait_tensor();
    __syncthreads();

    float8 acc[2][4] = {};
    int cur = 0;
    for (int kk = 0; kk < K; kk += 32) {
        if (kk + 32 < K) issue(cur ^ 1, kk + 32);

        const _Float16* As = &Abuf[cur][(wm * 32 + l16) * 32];
        half16 af[2];
        {
            const half8* p0 = (const half8*)(As + 8 * hi);
            af[0] = cat8(p0[0], p0[2]);
            const half8* p1 = (const half8*)(As + 16 * 32 + 8 * hi);
            af[1] = cat8(p1[0], p1[2]);
        }
        const _Float16* Bs = &Bbuf[cur][(wn * 64 + l16) * 32];
        half16 bf[4];
#pragma unroll
        for (int nt = 0; nt < 4; ++nt)
            bf[nt] = cat8(*(const half8*)(Bs + nt * 16 * 32 + 8 * hi), *(const half8*)(Bs + nt * 16 * 32 + 16 + 8 * hi));
#pragma unroll
        for (int mt = 0; mt < 2; ++mt)
#pragma unroll
            for (int nt = 0; nt < 4; ++nt)
                acc[mt][nt] = wmma_f32(af[mt], bf[nt], acc[mt][nt]);

        wait_async();
        wait_tensor();
        __syncthreads();
        cur ^= 1;
    }

    // C/D layout: lane holds col = l16, VGPR r holds row r + 8*hi
#pragma unroll
    for (int mt = 0; mt < 2; ++mt)
#pragma unroll
        for (int nt = 0; nt < 4; ++nt) {
            int col = colBase + wn * 64 + nt * 16 + l16;
#pragma unroll
            for (int r = 0; r < 8; ++r) {
                int row = rowBase + wm * 32 + mt * 16 + r + 8 * hi;
                C[(size_t)row * N + col] = acc[mt][nt][r];
            }
        }
}

// ---------------- repack qkv f32 [8192,1536] -> q16,k16 [(bh),s,64], vT16 [(bh),64,s] ----------------
__global__ void repack_qkv_kernel(const float* __restrict__ qkv,
                                  _Float16* __restrict__ q16,
                                  _Float16* __restrict__ k16,
                                  _Float16* __restrict__ vT16) {
    int idx = blockIdx.x * blockDim.x + threadIdx.x;      // 8192*512 threads
    int t = idx >> 9;            // token
    int hd = idx & 511;
    int h = hd >> 6, d = hd & 63;
    int b = t >> 12, s = t & 4095;
    const float* base = qkv + (size_t)t * 1536 + h * 192;
    float qv = base[d];
    float kv = base[64 + d];
    float vv = base[128 + d];
    size_t bh = (size_t)(b * 8 + h);
    q16[(bh * 4096 + s) * 64 + d] = (_Float16)qv;
    k16[(bh * 4096 + s) * 64 + d] = (_Float16)kv;
    vT16[(bh * 64 + d) * 4096 + s] = (_Float16)vv;
}

// ---------------- flash attention with async-staged K/V tiles ----------------
// grid (S/64, B*H), block 128 (4 waves); wave = 16 query rows; key tile = 32.
// All 4 waves share each K/V tile -> cooperative async copy into LDS.
__global__ __launch_bounds__(128) void flash_attn_kernel(const _Float16* __restrict__ q16,
                                                         const _Float16* __restrict__ k16,
                                                         const _Float16* __restrict__ vT16,
                                                         _Float16* __restrict__ o16) {
    const int S = 4096, Dh = 64;
    int bh = blockIdx.y;
    int tid = threadIdx.x;
    int wave = tid >> 5, lane = tid & 31;
    int l16 = lane & 15, hi = lane >> 4;
    int qRow0 = blockIdx.x * 64 + wave * 16;

    __shared__ __align__(64) _Float16 Kt[2][32 * 64];   // [key][dh]
    __shared__ __align__(64) _Float16 Vt[2][64 * 32];   // [dv][key]
    __shared__ __align__(64) _Float16 Pb[4][16 * 32];   // per-wave P
    _Float16* plds = Pb[wave];

    const _Float16* qbase = q16 + (size_t)bh * S * Dh;
    const _Float16* kg = k16 + (size_t)bh * S * Dh;
    const _Float16* vg = vT16 + (size_t)bh * Dh * S;

    // cooperative copy mapping (128 threads):
    int kRow = tid >> 2, kOff = (tid & 3) * 16;   // K tile 32x64 halfs
    int vRow = tid >> 1, vOff = (tid & 1) * 16;   // V tile 64x32 halfs

    auto issue = [&](int buf, int j) {
        const _Float16* kp = kg + (size_t)(j + kRow) * Dh + kOff;
        _Float16* kl = &Kt[buf][kRow * 64 + kOff];
        async_b128(kp,     kl);
        async_b128(kp + 8, kl + 8);
        const _Float16* vp = vg + (size_t)vRow * S + j + vOff;
        _Float16* vl = &Vt[buf][vRow * 32 + vOff];
        async_b128(vp,     vl);
        async_b128(vp + 8, vl + 8);
    };

    // Q A-fragments (k = 0..31, 32..63), loaded once from global
    half16 qa[2];
    {
        const _Float16* qr = qbase + (size_t)(qRow0 + l16) * Dh;
#pragma unroll
        for (int t = 0; t < 2; ++t) {
            const half8* p = (const half8*)(qr + 32 * t + 8 * hi);
            qa[t] = cat8(p[0], p[2]);
        }
    }

    // all-ones B fragment: D = P @ ones gives per-row sums in C-layout rows
    half16 onesb;
#pragma unroll
    for (int i = 0; i < 16; ++i) onesb[i] = (_Float16)1.0f;

    float m[8], l[8], alpha[8];
    float8 o[4] = {};
#pragma unroll
    for (int r = 0; r < 8; ++r) { m[r] = -3.0e38f; l[r] = 0.f; }

    const float scale = 0.044194173824159216f;   // 1/sqrt(512)

    issue(0, 0);
    wait_async();
    __syncthreads();

    int cur = 0;
    for (int j = 0; j < S; j += 32) {
        if (j + 32 < S) issue(cur ^ 1, j + 32);

        // S tiles: 16 queries x 32 keys, K-frags batched before the WMMA batch
        const _Float16* kb = Kt[cur];
        half16 kf[4];
        kf[0] = cat8(*(const half8*)(kb + l16 * 64 + 8 * hi), *(const half8*)(kb + l16 * 64 + 16 + 8 * hi));
        kf[1] = cat8(*(const half8*)(kb + l16 * 64 + 32 + 8 * hi), *(const half8*)(kb + l16 * 64 + 48 + 8 * hi));
        kf[2] = cat8(*(const half8*)(kb + (16 + l16) * 64 + 8 * hi), *(const half8*)(kb + (16 + l16) * 64 + 16 + 8 * hi));
        kf[3] = cat8(*(const half8*)(kb + (16 + l16) * 64 + 32 + 8 * hi), *(const half8*)(kb + (16 + l16) * 64 + 48 + 8 * hi));
        float8 s0 = {}, s1 = {};
        s0 = wmma_f32(qa[0], kf[0], s0);
        s0 = wmma_f32(qa[1], kf[1], s0);
        s1 = wmma_f32(qa[0], kf[2], s1);
        s1 = wmma_f32(qa[1], kf[3], s1);

        // online softmax: row max via wave32 shuffles; row sum via WMMA(P, ones)
#pragma unroll
        for (int r = 0; r < 8; ++r) {
            float v0 = s0[r] * scale, v1 = s1[r] * scale;
            float mx = fmaxf(v0, v1);
#pragma unroll
            for (int msk = 8; msk >= 1; msk >>= 1) mx = fmaxf(mx, __shfl_xor(mx, msk, 32));
            float mnew = fmaxf(m[r], mx);
            alpha[r] = __expf(m[r] - mnew);
            m[r] = mnew;
            float p0 = __expf(v0 - mnew);
            float p1 = __expf(v1 - mnew);
#pragma unroll
            for (int nt = 0; nt < 4; ++nt) o[nt][r] *= alpha[r];
            plds[(r + 8 * hi) * 32 + l16]      = (_Float16)p0;
            plds[(r + 8 * hi) * 32 + 16 + l16] = (_Float16)p1;
        }

        // P A-fragment (16x32) from per-wave LDS (wave-synchronous; DScnt waits inserted)
        half16 pf;
        {
            const half8* pp = (const half8*)(plds + l16 * 32 + 8 * hi);
            pf = cat8(pp[0], pp[2]);
        }

        // V fragments batched, then row-sum + O-update WMMAs
        half16 bv[4];
#pragma unroll
        for (int nt = 0; nt < 4; ++nt)
            bv[nt] = cat8(*(const half8*)(&Vt[cur][(nt * 16 + l16) * 32 + 8 * hi]), *(const half8*)(&Vt[cur][(nt * 16 + l16) * 32 + 16 + 8 * hi]));

        float8 rsum = {};
        rsum = wmma_f32(pf, onesb, rsum);
#pragma unroll
        for (int nt = 0; nt < 4; ++nt)
            o[nt] = wmma_f32(pf, bv[nt], o[nt]);
#pragma unroll
        for (int r = 0; r < 8; ++r) l[r] = l[r] * alpha[r] + rsum[r];

        wait_async();
        __syncthreads();
        cur ^= 1;
    }

    // epilogue: normalize and scatter into o16[8192, 512] with col = h*64 + dv
    int b = bh >> 3;
    int hcol = (bh & 7) * 64;
#pragma unroll
    for (int nt = 0; nt < 4; ++nt) {
        int col = hcol + nt * 16 + l16;
#pragma unroll
        for (int r = 0; r < 8; ++r) {
            int s = qRow0 + r + 8 * hi;
            float val = o[nt][r] / l[r];
            o16[((size_t)(b * 4096 + s)) * 512 + col] = (_Float16)val;
        }
    }
}

// ---------------- x1 = attn + b_out + x; f = silu(LN(x1)) -> f16 ----------------
__global__ __launch_bounds__(256) void bias_res_ln_silu_kernel(const float* __restrict__ attn,
                                                               const float* __restrict__ b_out,
                                                               const float* __restrict__ x,
                                                               const float* __restrict__ g,
                                                               const float* __restrict__ bb,
                                                               float* __restrict__ x1,
                                                               _Float16* __restrict__ f16o, int M) {
    int row  = blockIdx.x * 8 + (threadIdx.x >> 5);
    int lane = threadIdx.x & 31;
    if (row >= M) return;
    size_t base = (size_t)row * 512;
    float v[16];
    float s = 0.f;
#pragma unroll
    for (int i = 0; i < 16; ++i) {
        int c = lane + 32 * i;
        float t = attn[base + c] + b_out[c] + x[base + c];
        v[i] = t;
        x1[base + c] = t;
        s += t;
    }
#pragma unroll
    for (int m = 16; m >= 1; m >>= 1) s += __shfl_xor(s, m, 32);
    float mu = s * (1.f / 512.f);
    float var = 0.f;
#pragma unroll
    for (int i = 0; i < 16; ++i) { float d = v[i] - mu; var += d * d; }
#pragma unroll
    for (int m = 16; m >= 1; m >>= 1) var += __shfl_xor(var, m, 32);
    float rstd = rsqrtf(var * (1.f / 512.f) + 1e-5f);
#pragma unroll
    for (int i = 0; i < 16; ++i) {
        int c = lane + 32 * i;
        float y = (v[i] - mu) * rstd * g[c] + bb[c];
        float sy = y / (1.f + __expf(-y));
        f16o[base + c] = (_Float16)sy;
    }
}

// ---------------- generic row LN + SiLU -> f16, row length R ----------------
template <int R>
__global__ __launch_bounds__(256) void ln_silu_kernel(const float* __restrict__ src,
                                                      const float* __restrict__ g,
                                                      const float* __restrict__ bb,
                                                      _Float16* __restrict__ dst, int M) {
    constexpr int PER = R / 32;
    int row  = blockIdx.x * 8 + (threadIdx.x >> 5);
    int lane = threadIdx.x & 31;
    if (row >= M) return;
    size_t base = (size_t)row * R;
    float v[PER];
    float s = 0.f;
#pragma unroll
    for (int i = 0; i < PER; ++i) { v[i] = src[base + lane + 32 * i]; s += v[i]; }
#pragma unroll
    for (int m = 16; m >= 1; m >>= 1) s += __shfl_xor(s, m, 32);
    float mu = s * (1.f / R);
    float var = 0.f;
#pragma unroll
    for (int i = 0; i < PER; ++i) { float d = v[i] - mu; var += d * d; }
#pragma unroll
    for (int m = 16; m >= 1; m >>= 1) var += __shfl_xor(var, m, 32);
    float rstd = rsqrtf(var * (1.f / R) + 1e-5f);
#pragma unroll
    for (int i = 0; i < PER; ++i) {
        int c = lane + 32 * i;
        float y = (v[i] - mu) * rstd * g[c] + bb[c];
        float sy = y / (1.f + __expf(-y));
        dst[base + c] = (_Float16)sy;
    }
}

// ---------------- out = x1 + f5 ----------------
__global__ void final_add_kernel(const float* __restrict__ x1, const float* __restrict__ f5,
                                 float* __restrict__ out, int n) {
    int i = blockIdx.x * blockDim.x + threadIdx.x;
    if (i < n) out[i] = x1[i] + f5[i];
}

extern "C" void kernel_launch(void* const* d_in, const int* in_sizes, int n_in,
                              void* d_out, int out_size, void* d_ws, size_t ws_size,
                              hipStream_t stream) {
    const float* x       = (const float*)d_in[0];
    const float* norm1_g = (const float*)d_in[1];
    const float* norm1_b = (const float*)d_in[2];
    const float* w_qkv   = (const float*)d_in[3];
    const float* w_out   = (const float*)d_in[4];
    const float* b_out   = (const float*)d_in[5];
    const float* ln0_g   = (const float*)d_in[6];
    const float* ln0_b   = (const float*)d_in[7];
    const float* w2      = (const float*)d_in[8];
    const float* ln3_g   = (const float*)d_in[9];
    const float* ln3_b   = (const float*)d_in[10];
    const float* w5      = (const float*)d_in[11];
    float* out = (float*)d_out;

    const int M = 8192;   // B*S

    char* ws = (char*)d_ws;
    size_t off = 0;
    auto take = [&](size_t bytes) -> void* {
        void* p = ws + off;
        off = (off + bytes + 255) & ~(size_t)255;
        return p;
    };
    _Float16* h16     = (_Float16*)take((size_t)M * 512 * 2);
    _Float16* wqkv16  = (_Float16*)take((size_t)1536 * 512 * 2);
    _Float16* wout16  = (_Float16*)take((size_t)512 * 512 * 2);
    _Float16* w2_16   = (_Float16*)take((size_t)1024 * 512 * 2);
    _Float16* w5_16   = (_Float16*)take((size_t)512 * 1024 * 2);
    float*    qkv     = (float*)   take((size_t)M * 1536 * 4);
    _Float16* q16     = (_Float16*)take((size_t)16 * 4096 * 64 * 2);
    _Float16* k16     = (_Float16*)take((size_t)16 * 4096 * 64 * 2);
    _Float16* vT16    = (_Float16*)take((size_t)16 * 64 * 4096 * 2);
    _Float16* o16     = (_Float16*)take((size_t)M * 512 * 2);
    float*    attn    = (float*)   take((size_t)M * 512 * 4);
    float*    x1      = (float*)   take((size_t)M * 512 * 4);
    _Float16* f16buf  = (_Float16*)take((size_t)M * 512 * 2);
    float*    f2      = (float*)   take((size_t)M * 1024 * 4);
    _Float16* g16     = (_Float16*)take((size_t)M * 1024 * 2);
    float*    f5      = (float*)   take((size_t)M * 512 * 4);

    // 1) weight conversions to fp16
    f32_to_f16_kernel<<<(1536 * 512 + 255) / 256, 256, 0, stream>>>(w_qkv, wqkv16, 1536 * 512);
    f32_to_f16_kernel<<<(512 * 512 + 255) / 256, 256, 0, stream>>>(w_out, wout16, 512 * 512);
    f32_to_f16_kernel<<<(1024 * 512 + 255) / 256, 256, 0, stream>>>(w2, w2_16, 1024 * 512);
    f32_to_f16_kernel<<<(512 * 1024 + 255) / 256, 256, 0, stream>>>(w5, w5_16, 512 * 1024);

    // 2) h = LN(x) -> f16
    ln_x_kernel<<<M / 8, 256, 0, stream>>>(x, norm1_g, norm1_b, h16, M);

    // 3) qkv = h @ w_qkv^T   (8192 x 1536, K=512)
    wmma_gemm_kernel<512><<<dim3(M / 128, 1536 / 128), 256, 0, stream>>>(h16, wqkv16, qkv, 1536);

    // 4) split + transpose V
    repack_qkv_kernel<<<(M * 512) / 256, 256, 0, stream>>>(qkv, q16, k16, vT16);

    // 5) flash attention -> o16 [8192, 512]
    flash_attn_kernel<<<dim3(4096 / 64, 16), 128, 0, stream>>>(q16, k16, vT16, o16);

    // 6) attn = o @ w_out^T  (8192 x 512, K=512)
    wmma_gemm_kernel<512><<<dim3(M / 128, 512 / 128), 256, 0, stream>>>(o16, wout16, attn, 512);

    // 7) x1 = attn + b_out + x ; f = silu(LN(x1)) -> f16
    bias_res_ln_silu_kernel<<<M / 8, 256, 0, stream>>>(attn, b_out, x, ln0_g, ln0_b, x1, f16buf, M);

    // 8) f2 = f @ w2^T  (8192 x 1024, K=512)
    wmma_gemm_kernel<512><<<dim3(M / 128, 1024 / 128), 256, 0, stream>>>(f16buf, w2_16, f2, 1024);

    // 9) g = silu(LN(f2)) -> f16  (rows of 1024)
    ln_silu_kernel<1024><<<M / 8, 256, 0, stream>>>(f2, ln3_g, ln3_b, g16, M);

    // 10) f5 = g @ w5^T  (8192 x 512, K=1024)
    wmma_gemm_kernel<1024><<<dim3(M / 128, 512 / 128), 256, 0, stream>>>(g16, w5_16, f5, 512);

    // 11) out = x1 + f5
    final_add_kernel<<<(M * 512) / 256, 256, 0, stream>>>(x1, f5, out, M * 512);
}
